// VectorQuantizer_56745107914922
// MI455X (gfx1250) — matrix operand order repaired and checked by reading, not compiled, not run on hardware
//
#include <hip/hip_runtime.h>
#include <hip/hip_bf16.h>

typedef __attribute__((ext_vector_type(16))) _Float16 v16h;
typedef __attribute__((ext_vector_type(8)))  float    v8f;

#define NUM_EMB    256
#define EMB_DIM    32
#define NUM_GROUPS 16
#define TOTAL_DIM  512
#define N_ROWS     65536        // 16 * 4096
#define ROW_TILES  4096         // N_ROWS / 16
#define QSTE_ELEMS (N_ROWS * TOTAL_DIM)          // 33554432
#define LOSS_OFF   ((size_t)QSTE_ELEMS)          // 33554432
#define IDX_OFF    (LOSS_OFF + 1)                // 33554433

// ---------------------------------------------------------------------------
// Kernel 1: codebook prep.
//   ef16[g][k][d]    = (f16)embeds[g][k][d]   (natural d order: B fragment
//                      lane-half h reads d = h*16 .. h*16+15 contiguously)
//   ebiasT[g][c][t]  = -0.5*||e[g][t*16+c]||^2  (transposed so lane c grabs
//                      all 16 tile biases with 4 contiguous float4 loads)
// ---------------------------------------------------------------------------
__global__ __launch_bounds__(256) void vq_pre(const float* __restrict__ embeds,
                                              _Float16* __restrict__ ef16,
                                              float* __restrict__ ebiasT) {
    int id = blockIdx.x * 256 + threadIdx.x;          // 0 .. 4095 (g*256 + k)
    if (id >= NUM_GROUPS * NUM_EMB) return;
    const int g = id >> 8;
    const int k = id & 255;
    const float* e = embeds + (size_t)id * EMB_DIM;
    float v[EMB_DIM];
    float ss = 0.0f;
#pragma unroll
    for (int d = 0; d < EMB_DIM; ++d) { v[d] = e[d]; ss += v[d] * v[d]; }
    // transposed bias: col = k & 15, tile = k >> 4
    ebiasT[g * NUM_EMB + (k & 15) * 16 + (k >> 4)] = -0.5f * ss;
    _Float16* o = ef16 + (size_t)id * EMB_DIM;
#pragma unroll
    for (int d = 0; d < EMB_DIM; ++d) o[d] = (_Float16)v[d];
}

// ---------------------------------------------------------------------------
// Kernel 2: main VQ. One block = 16 rows; wave w handles groups w and w+8.
// Scores via v_wmma_f32_16x16x32_f16 (A = normalized rows f16, B = codebook
// f16, C preloaded with -0.5*||e||^2 from registers). Two-accumulator
// pipeline: WMMA for tile t issues while tile t-1's argmax compares
// co-execute. __launch_bounds__(256, 4) caps VGPRs so >=4 waves/SIMD stay
// resident to hide HBM latency in the streaming phases. Exact f32 epilogue.
// ---------------------------------------------------------------------------
__global__ __launch_bounds__(256, 4) void vq_main(const float* __restrict__ x,
                                                  const float* __restrict__ embeds,
                                                  const _Float16* __restrict__ ef16,
                                                  const float* __restrict__ ebiasT,
                                                  float* __restrict__ out,
                                                  float* __restrict__ partials) {
    __shared__ float wsum[8];
    const int tid  = threadIdx.x;
    const int wave = tid >> 5;
    const int lane = tid & 31;
    const int hf   = lane >> 4;       // lane half (A/B/C layout halves)
    const int l15  = lane & 15;
    const int rt   = blockIdx.x;      // row tile
    const int row  = rt * 16 + l15;   // this lane's matrix row (M = l15)

    float lossAcc = 0.0f;

    for (int gi = 0; gi < 2; ++gi) {
        const int g = wave + gi * 8;

        // ---- all 16 tile biases for this lane's column, 4 float4 loads
        const float* bgT = ebiasT + g * NUM_EMB + l15 * 16;
        float4 bb0 = *(const float4*)(bgT);
        float4 bb1 = *(const float4*)(bgT + 4);
        float4 bb2 = *(const float4*)(bgT + 8);
        float4 bb3 = *(const float4*)(bgT + 12);
        float biasArr[16] = {bb0.x, bb0.y, bb0.z, bb0.w,
                             bb1.x, bb1.y, bb1.z, bb1.w,
                             bb2.x, bb2.y, bb2.z, bb2.w,
                             bb3.x, bb3.y, bb3.z, bb3.w};

        // ---- A fragment: this code's contiguous k order, not the ISA table order.
        // half 0 holds K in {0..15}; half 1 holds {16..31};
        // a[i] <- K = hf*16 + i (contiguous, matches B).
        const float* rowp = x + (size_t)row * TOTAL_DIM + g * EMB_DIM;
        const int d0 = hf * 16;
        const int d1 = hf * 16 + 8;
        float4 f0 = *(const float4*)(rowp + d0);
        float4 f1 = *(const float4*)(rowp + d0 + 4);
        float4 f2 = *(const float4*)(rowp + d1);
        float4 f3 = *(const float4*)(rowp + d1 + 4);
        float r16[16] = {f0.x, f0.y, f0.z, f0.w, f1.x, f1.y, f1.z, f1.w,
                         f2.x, f2.y, f2.z, f2.w, f3.x, f3.y, f3.z, f3.w};
        float ss = 0.0f;
#pragma unroll
        for (int i = 0; i < 16; ++i) ss += r16[i] * r16[i];
        ss += __shfl_xor(ss, 16, 32);             // other half of the row
        const float scale = __builtin_amdgcn_rcpf(sqrtf(ss) + 1e-6f);

        float xr[16];
        v16h a;
#pragma unroll
        for (int i = 0; i < 16; ++i) {
            xr[i] = r16[i] * scale;
            a[i]  = (_Float16)xr[i];
        }

        // ---- scan 256 codes in 16 WMMA tiles with 1-tile lookahead.
        float bestv[8];
        int   besti[8];
#pragma unroll
        for (int j = 0; j < 8; ++j) { bestv[j] = -3.4e38f; besti[j] = 0; }

        const _Float16* eg = ef16 + (size_t)g * NUM_EMB * EMB_DIM;

        v8f cc[2];
        {   // tile 0
            v16h b = *(const v16h*)(eg + (size_t)l15 * EMB_DIM + hf * 16);
            v8f c;
#pragma unroll
            for (int j = 0; j < 8; ++j) c[j] = biasArr[0];
            cc[0] = __builtin_amdgcn_wmma_f32_16x16x32_f16(
                        false, a, false, b, (short)0, c, false, false);
        }
#pragma unroll
        for (int t = 1; t < 16; ++t) {
            // issue WMMA for tile t into the alternate accumulator
            const int k = t * 16 + l15;
            v16h b = *(const v16h*)(eg + (size_t)k * EMB_DIM + hf * 16);
            v8f c;
#pragma unroll
            for (int j = 0; j < 8; ++j) c[j] = biasArr[t];
            cc[t & 1] = __builtin_amdgcn_wmma_f32_16x16x32_f16(
                            false, a, false, b, (short)0, c, false, false);
            // consume tile t-1 (co-executes with the WMMA above)
            const int kp = (t - 1) * 16 + l15;
#pragma unroll
            for (int j = 0; j < 8; ++j) {         // row hf*8+j, col kp
                float v = cc[(t - 1) & 1][j];
                if (v > bestv[j]) { bestv[j] = v; besti[j] = kp; }
            }
        }
        {   // consume tile 15
            const int kp = 15 * 16 + l15;
#pragma unroll
            for (int j = 0; j < 8; ++j) {
                float v = cc[1][j];
                if (v > bestv[j]) { bestv[j] = v; besti[j] = kp; }
            }
        }

        // ---- reduce argmax across the 16 lanes of each half (ties -> min k).
#pragma unroll
        for (int m = 1; m < 16; m <<= 1) {
#pragma unroll
            for (int j = 0; j < 8; ++j) {
                float ov = __shfl_xor(bestv[j], m, 32);
                int   oi = __shfl_xor(besti[j], m, 32);
                if (ov > bestv[j] || (ov == bestv[j] && oi < besti[j])) {
                    bestv[j] = ov; besti[j] = oi;
                }
            }
        }
        // exchange halves: otherI[j] = best idx for row (1-hf)*8 + j
        int otherI[8];
#pragma unroll
        for (int j = 0; j < 8; ++j) otherI[j] = __shfl_xor(besti[j], 16, 32);

        // pick the index for this lane's own row m = l15 (half m>>3, slot m&7)
        const int rsel    = l15 & 7;
        const int rowHalf = l15 >> 3;
        int myIdx = 0;
#pragma unroll
        for (int j = 0; j < 8; ++j)
            if (rsel == j) myIdx = (rowHalf == hf) ? besti[j] : otherI[j];

        if (hf == 0)
            out[IDX_OFF + (size_t)row * NUM_GROUPS + g] = (float)myIdx;

        // ---- exact f32 epilogue: quantized output + residual for the loss.
        const float* ep = embeds + ((size_t)g * NUM_EMB + myIdx) * EMB_DIM;
        float4 e0 = *(const float4*)(ep + d0);
        float4 e1 = *(const float4*)(ep + d0 + 4);
        float4 e2 = *(const float4*)(ep + d1);
        float4 e3 = *(const float4*)(ep + d1 + 4);
        float er[16] = {e0.x, e0.y, e0.z, e0.w, e1.x, e1.y, e1.z, e1.w,
                        e2.x, e2.y, e2.z, e2.w, e3.x, e3.y, e3.z, e3.w};
        float acc = 0.0f;
#pragma unroll
        for (int i = 0; i < 16; ++i) {
            float df = er[i] - xr[i];
            acc += df * df;
        }
        lossAcc += acc;

        float* op = out + (size_t)row * TOTAL_DIM + g * EMB_DIM;
        *(float4*)(op + d0)     = e0;
        *(float4*)(op + d0 + 4) = e1;
        *(float4*)(op + d1)     = e2;
        *(float4*)(op + d1 + 4) = e3;
    }

    // deterministic block partial sum
#pragma unroll
    for (int m = 16; m >= 1; m >>= 1) lossAcc += __shfl_xor(lossAcc, m, 32);
    if (lane == 0) wsum[wave] = lossAcc;
    __syncthreads();
    if (tid == 0) {
        float s = 0.0f;
#pragma unroll
        for (int w = 0; w < 8; ++w) s += wsum[w];
        partials[rt] = s;
    }
}

// ---------------------------------------------------------------------------
// Kernel 3: deterministic reduction of 4096 partials -> total_loss.
// total = (1 + BETA) * sum / (N_ROWS * TOTAL_DIM), BETA = 0.25.
// ---------------------------------------------------------------------------
__global__ __launch_bounds__(256) void vq_final(const float* __restrict__ partials,
                                                float* __restrict__ out) {
    __shared__ float s[256];
    const int t = threadIdx.x;
    float a = 0.0f;
    for (int i = t; i < ROW_TILES; i += 256) a += partials[i];
    s[t] = a;
    __syncthreads();
    for (int m = 128; m > 0; m >>= 1) {
        if (t < m) s[t] += s[t + m];
        __syncthreads();
    }
    if (t == 0)
        out[LOSS_OFF] = s[0] * (1.25f / ((float)N_ROWS * (float)TOTAL_DIM));
}

// ---------------------------------------------------------------------------
extern "C" void kernel_launch(void* const* d_in, const int* in_sizes, int n_in,
                              void* d_out, int out_size, void* d_ws, size_t ws_size,
                              hipStream_t stream) {
    const float* inputs = (const float*)d_in[0];   // (16, 4096, 512) f32
    const float* embeds = (const float*)d_in[1];   // (16, 256, 32) f32
    float* out = (float*)d_out;

    // workspace layout
    char* ws = (char*)d_ws;
    _Float16* ef16   = (_Float16*)(ws);                       // 262144 B
    float*    ebiasT = (float*)(ws + 262144);                 //  16384 B
    float*    parts  = (float*)(ws + 262144 + 16384);         //  16384 B

    vq_pre  <<<NUM_GROUPS * NUM_EMB / 256, 256, 0, stream>>>(embeds, ef16, ebiasT);
    vq_main <<<ROW_TILES, 256, 0, stream>>>(inputs, embeds, ef16, ebiasT, out, parts);
    vq_final<<<1, 256, 0, stream>>>(parts, out);
}
